// SelfAttention_51384988729912
// MI455X (gfx1250) — compile-verified, matrix operand order checked by reading, not run on hardware
//
#include <hip/hip_runtime.h>
#include <hip/hip_bf16.h>

// Problem constants (match reference)
#define BB 4
#define TT 2048
#define GG 1024
#define HH 16
#define HD 64

typedef __bf16 bf16_t;
typedef __attribute__((ext_vector_type(16))) __bf16 v16bf;
typedef __attribute__((ext_vector_type(8)))  float  v8f;
typedef __attribute__((ext_vector_type(4)))  unsigned int u32x4;
typedef __attribute__((ext_vector_type(4)))  int i32x4;

// CDNA5 async global->LDS copy (ASYNCcnt-tracked) if the toolchain declares it.
#if defined(__gfx1250__) && __has_builtin(__builtin_amdgcn_global_load_async_to_lds_b128) && __has_builtin(__builtin_amdgcn_s_wait_asynccnt)
#define HAVE_ASYNC_LDS 1
#endif

union FragU { v16bf v; u32x4 u[2]; };

// Load a 16x32 bf16 fragment (A layout; B uses identical layout when the
// operand is stored [n][k] row-major).  Per CDNA5 ISA 7.12.2:
//   lanes 0-15 : row = lane,  K chunks {0..7, 16..23}
//   lanes 16-31: row = lane-16, K chunks {8..15, 24..31}
// => two contiguous 16-byte loads per lane.
__device__ __forceinline__ v16bf load_frag(const bf16_t* base, int stride, int lane) {
    const int row = lane & 15;
    const int c0  = (lane < 16) ? 0 : 8;
    const bf16_t* p = base + (size_t)row * stride + c0;
    FragU f;
    f.u[0] = *(const u32x4*)(p);
    f.u[1] = *(const u32x4*)(p + 16);
    return f.v;
}

__device__ __forceinline__ v8f wmma_bf16(v16bf a, v16bf b, v8f c) {
    return __builtin_amdgcn_wmma_f32_16x16x32_bf16(
        /*neg_a=*/false, a, /*neg_b=*/false, b,
        /*c_mod=*/(short)0, c, /*reuse_a=*/false, /*reuse_b=*/false);
}

__device__ __forceinline__ v8f zero_v8f() {
    v8f z;
    for (int i = 0; i < 8; ++i) z[i] = 0.0f;
    return z;
}

// Copy 16 bytes global -> LDS (async path uses GLOBAL_LOAD_ASYNC_TO_LDS_B128).
__device__ __forceinline__ void stage16(const bf16_t* g, bf16_t* l) {
#ifdef HAVE_ASYNC_LDS
    __builtin_amdgcn_global_load_async_to_lds_b128(
        (__attribute__((address_space(1))) i32x4*)g,
        (__attribute__((address_space(3))) i32x4*)l,
        /*offset=*/0, /*cpol=*/0);
#else
    *(u32x4*)l = *(const u32x4*)g;
#endif
}

__device__ __forceinline__ void stage_fence() {
#ifdef HAVE_ASYNC_LDS
    __builtin_amdgcn_s_wait_asynccnt(0);
#endif
}

// ---------------- fp32 -> bf16 conversions ----------------
__global__ void cvt_bf16_k(const float* __restrict__ src, bf16_t* __restrict__ dst, int n) {
    int i = blockIdx.x * 256 + threadIdx.x;
    if (i < n) dst[i] = (bf16_t)src[i];
}

// w is [K=G][N=G] row-major; write wt[n][k] = w[k][n]  (bf16, transposed)
__global__ void transpose_cvt_k(const float* __restrict__ w, bf16_t* __restrict__ wt) {
    int idx = blockIdx.x * 256 + threadIdx.x;      // over G*G, row-major in wt
    int n = idx >> 10;                             // idx / G
    int k = idx & (GG - 1);
    wt[idx] = (bf16_t)w[(size_t)k * GG + n];
}

// ---------------- bf16 GEMM:  C[M,N] = A[M,K] * Bt[N,K]^T + bias ----------------
// MODE 0: store Q  bf16 [B,H,T,HD]
// MODE 1: store K  bf16 [B,H,T,HD]
// MODE 2: store Vt bf16 [B,H,HD,T]   (transposed for P*V B-fragments)
// MODE 3: store fp32 [M,G] (final output)
template <int MODE>
__global__ __launch_bounds__(256) void gemm_bf16_k(const bf16_t* __restrict__ A,
                                                   const bf16_t* __restrict__ Bt,
                                                   const float*  __restrict__ bias,
                                                   void* __restrict__ outp) {
    __shared__ __align__(16) bf16_t As[128 * 32];
    __shared__ __align__(16) bf16_t Bs[64 * 32];

    const int tid  = threadIdx.x;
    const int bm   = blockIdx.y * 128;
    const int bn   = blockIdx.x * 64;
    const int wave = tid >> 5;
    const int lane = tid & 31;
    const int wm   = (wave >> 1) * 32;   // 4 waves along M
    const int wn   = (wave & 1) * 32;    // 2 waves along N

    v8f acc[2][2];
    for (int i = 0; i < 2; ++i)
        for (int j = 0; j < 2; ++j) acc[i][j] = zero_v8f();

    const int ar = tid >> 1, ac = (tid & 1) * 16;     // A tile mapping
    const int br = tid >> 2, bc = (tid & 3) * 8;      // B tile mapping

    for (int k0 = 0; k0 < GG; k0 += 32) {
        {   // A tile: 128 rows x 32 cols, 16 elems (32B) per thread
            const u32x4* src = (const u32x4*)(A + (size_t)(bm + ar) * GG + k0 + ac);
            u32x4* dst = (u32x4*)(As + ar * 32 + ac);
            dst[0] = src[0];
            dst[1] = src[1];
        }
        {   // B tile (Wt, [n][k]): 64 rows x 32 cols, 8 elems per thread
            *(u32x4*)(Bs + br * 32 + bc) =
                *(const u32x4*)(Bt + (size_t)(bn + br) * GG + k0 + bc);
        }
        if (k0 + 32 < GG) {   // hint next K tile into cache (global_prefetch_b8)
            __builtin_prefetch(A  + (size_t)(bm + ar) * GG + k0 + 32 + ac, 0, 0);
            __builtin_prefetch(Bt + (size_t)(bn + br) * GG + k0 + 32 + bc, 0, 0);
        }
        __syncthreads();

        v16bf af0 = load_frag(As + (wm + 0)  * 32, 32, lane);
        v16bf af1 = load_frag(As + (wm + 16) * 32, 32, lane);
        v16bf bf0 = load_frag(Bs + (wn + 0)  * 32, 32, lane);
        v16bf bf1 = load_frag(Bs + (wn + 16) * 32, 32, lane);

        acc[0][0] = wmma_bf16(af0, bf0, acc[0][0]);
        acc[0][1] = wmma_bf16(af0, bf1, acc[0][1]);
        acc[1][0] = wmma_bf16(af1, bf0, acc[1][0]);
        acc[1][1] = wmma_bf16(af1, bf1, acc[1][1]);

        __syncthreads();
    }

    // Epilogue.  C layout: VGPR r -> row = r + (lane<16 ? 0 : 8); col = lane&15.
    const int rbase = (lane < 16) ? 0 : 8;
    for (int mi = 0; mi < 2; ++mi) {
        for (int ni = 0; ni < 2; ++ni) {
            const int n  = bn + wn + ni * 16 + (lane & 15);
            const float bb = bias[n];
            for (int r = 0; r < 8; ++r) {
                const int m = bm + wm + mi * 16 + r + rbase;
                const float val = acc[mi][ni][r] + bb;
                if (MODE == 3) {
                    ((float*)outp)[(size_t)m * GG + n] = val;
                } else {
                    const int b = m >> 11;           // m / T
                    const int t = m & (TT - 1);
                    const int h = n >> 6;            // n / HD
                    const int d = n & (HD - 1);
                    const bf16_t vv = (bf16_t)val;
                    if (MODE == 2)
                        ((bf16_t*)outp)[(((size_t)(b * HH + h)) * HD + d) * TT + t] = vv;
                    else
                        ((bf16_t*)outp)[(((size_t)(b * HH + h)) * TT + t) * HD + d] = vv;
                }
            }
        }
    }
}

// ---------------- Flash attention ----------------
// grid = (T/64, B*H), block = 128 (4 waves).  Wave w owns 16 q-rows.
// Per key block of 64: cooperatively stage K[64][64] and Vt[64][64] tiles into
// LDS (async-to-LDS on CDNA5), S = Q*K^T via WMMA, online softmax in LDS,
// O += P*V via WMMA.
__global__ __launch_bounds__(128) void attn_k(const bf16_t* __restrict__ Q,
                                              const bf16_t* __restrict__ K,
                                              const bf16_t* __restrict__ Vt,
                                              bf16_t* __restrict__ Y) {
    __shared__ __align__(16) float  Sbuf[4][16][64];
    __shared__ __align__(16) bf16_t Pbuf[4][16][64];
    __shared__ __align__(16) bf16_t Ks[64 * 64];   // [key][d]
    __shared__ __align__(16) bf16_t Vs[64 * 64];   // [d][key]
    __shared__ float fac[4][16];
    __shared__ float linv[4][16];

    const int bh   = blockIdx.y;
    const int b    = bh >> 4;            // / H
    const int h    = bh & (HH - 1);
    const int qblk = blockIdx.x;
    const int wave = threadIdx.x >> 5;
    const int lane = threadIdx.x & 31;
    const int tid  = threadIdx.x;
    const int qrow = qblk * 64 + wave * 16;
    const int rbase = (lane < 16) ? 0 : 8;

    const bf16_t* Qh = Q  + (size_t)bh * TT * HD;
    const bf16_t* Kh = K  + (size_t)bh * TT * HD;
    const bf16_t* Vh = Vt + (size_t)bh * HD * TT;

    // Q fragments for this wave's 16 rows (d = 0..31 and 32..63)
    v16bf aq0, aq1;
    {
        const bf16_t* p = Qh + (size_t)qrow * HD;
        aq0 = load_frag(p,      HD, lane);
        aq1 = load_frag(p + 32, HD, lane);
    }

    v8f o[4];
    for (int nt = 0; nt < 4; ++nt) o[nt] = zero_v8f();
    float m_run = -1e30f, l_run = 0.0f;     // valid in lanes 0..15 (one row each)
    const float scale = 0.125f;             // 1/sqrt(64)
    const int nkb = qblk + 1;               // uniform trip count across the block

    for (int kb = 0; kb < nkb; ++kb) {
        const int key0 = kb * 64;

        // ---- stage K and Vt tiles into LDS (8KB each, 16B chunks) ----
        // 64x64 bf16 tile = 512 chunks of 16B; 128 threads * 4 chunks.
        for (int i = 0; i < 4; ++i) {
            const int c   = i * 128 + tid;    // 0..511
            const int row = c >> 3;           // 0..63
            const int col = (c & 7) * 8;      // 0..56
            stage16(Kh + (size_t)(key0 + row) * HD + col, Ks + row * 64 + col);
        }
        for (int i = 0; i < 4; ++i) {
            const int c   = i * 128 + tid;
            const int row = c >> 3;
            const int col = (c & 7) * 8;
            stage16(Vh + (size_t)row * TT + key0 + col, Vs + row * 64 + col);
        }
        stage_fence();
        __syncthreads();

        // ---- S = Q * K^T  (16 q-rows x 64 keys), K tiles from LDS ----
        for (int nt = 0; nt < 4; ++nt) {
            v16bf bk0 = load_frag(Ks + (nt * 16) * 64,      64, lane);
            v16bf bk1 = load_frag(Ks + (nt * 16) * 64 + 32, 64, lane);
            v8f s = zero_v8f();
            s = wmma_bf16(aq0, bk0, s);
            s = wmma_bf16(aq1, bk1, s);
            for (int r = 0; r < 8; ++r)
                Sbuf[wave][rbase + r][nt * 16 + (lane & 15)] = s[r];
        }
        __syncthreads();

        // ---- online softmax, one row per lane (lanes 0..15) ----
        if (lane < 16) {
            const int row = lane;
            const int qt  = qrow + row;
            float bmax = -1e30f;
            for (int j = 0; j < 64; ++j) {
                float sv = Sbuf[wave][row][j] * scale;
                if (key0 + j > qt) sv = -1e30f;      // causal mask
                Sbuf[wave][row][j] = sv;
                bmax = fmaxf(bmax, sv);
            }
            const float mnew = fmaxf(m_run, bmax);
            const float f    = __expf(m_run - mnew);
            float lsum = 0.0f;
            for (int j = 0; j < 64; ++j) {
                const float p = __expf(Sbuf[wave][row][j] - mnew);
                lsum += p;
                Pbuf[wave][row][j] = (bf16_t)p;
            }
            l_run = l_run * f + lsum;
            m_run = mnew;
            fac[wave][row] = f;
        }
        __syncthreads();

        // ---- rescale O, then O += P * V (V tiles from LDS) ----
        float fr[8];
        for (int r = 0; r < 8; ++r) fr[r] = fac[wave][rbase + r];
        for (int nt = 0; nt < 4; ++nt)
            for (int r = 0; r < 8; ++r) o[nt][r] *= fr[r];

        v16bf ap0 = load_frag(&Pbuf[wave][0][0],      64, lane);   // keys 0..31
        v16bf ap1 = load_frag(&Pbuf[wave][0][0] + 32, 64, lane);   // keys 32..63
        for (int nt = 0; nt < 4; ++nt) {
            v16bf bv0 = load_frag(Vs + (nt * 16) * 64,      64, lane);
            v16bf bv1 = load_frag(Vs + (nt * 16) * 64 + 32, 64, lane);
            o[nt] = wmma_bf16(ap0, bv0, o[nt]);
            o[nt] = wmma_bf16(ap1, bv1, o[nt]);
        }
        __syncthreads();   // guard Ks/Vs/Sbuf/Pbuf reuse next iteration
    }

    if (lane < 16) linv[wave][lane] = 1.0f / l_run;
    __syncthreads();

    float lr[8];
    for (int r = 0; r < 8; ++r) lr[r] = linv[wave][rbase + r];
    for (int nt = 0; nt < 4; ++nt) {
        const int d = nt * 16 + (lane & 15);
        for (int r = 0; r < 8; ++r) {
            const int t = qrow + rbase + r;
            Y[((size_t)b * TT + t) * GG + h * HD + d] = (bf16_t)(o[nt][r] * lr[r]);
        }
    }
}

// ---------------- host launcher ----------------
extern "C" void kernel_launch(void* const* d_in, const int* in_sizes, int n_in,
                              void* d_out, int out_size, void* d_ws, size_t ws_size,
                              hipStream_t stream) {
    const float* x  = (const float*)d_in[0];
    const float* wq = (const float*)d_in[1];
    const float* bq = (const float*)d_in[2];
    const float* wk = (const float*)d_in[3];
    const float* bk = (const float*)d_in[4];
    const float* wv = (const float*)d_in[5];
    const float* bv = (const float*)d_in[6];
    const float* wp = (const float*)d_in[7];
    const float* bp = (const float*)d_in[8];

    char* ws = (char*)d_ws;
    const size_t MB = (size_t)1 << 20;
    bf16_t* xb  = (bf16_t*)(ws + 0);        // 16 MB  [B*T, G]
    bf16_t* wqT = (bf16_t*)(ws + 16 * MB);  //  2 MB  [N,K]
    bf16_t* wkT = (bf16_t*)(ws + 18 * MB);
    bf16_t* wvT = (bf16_t*)(ws + 20 * MB);
    bf16_t* wpT = (bf16_t*)(ws + 22 * MB);
    bf16_t* Qb  = (bf16_t*)(ws + 24 * MB);  // 16 MB  [B,H,T,HD]
    bf16_t* Kb  = (bf16_t*)(ws + 40 * MB);  // 16 MB  [B,H,T,HD]
    bf16_t* Vtb = (bf16_t*)(ws + 56 * MB);  // 16 MB  [B,H,HD,T]
    bf16_t* Yb  = (bf16_t*)(ws + 72 * MB);  // 16 MB  [B*T, G]

    const int nx = BB * TT * GG;            // 8M
    const int nw = GG * GG;                 // 1M
    cvt_bf16_k<<<nx / 256, 256, 0, stream>>>(x, xb, nx);
    transpose_cvt_k<<<nw / 256, 256, 0, stream>>>(wq, wqT);
    transpose_cvt_k<<<nw / 256, 256, 0, stream>>>(wk, wkT);
    transpose_cvt_k<<<nw / 256, 256, 0, stream>>>(wv, wvT);
    transpose_cvt_k<<<nw / 256, 256, 0, stream>>>(wp, wpT);

    dim3 ggrid(GG / 64, (BB * TT) / 128);   // (16, 64)
    gemm_bf16_k<0><<<ggrid, 256, 0, stream>>>(xb, wqT, bq, (void*)Qb);
    gemm_bf16_k<1><<<ggrid, 256, 0, stream>>>(xb, wkT, bk, (void*)Kb);
    gemm_bf16_k<2><<<ggrid, 256, 0, stream>>>(xb, wvT, bv, (void*)Vtb);

    attn_k<<<dim3(TT / 64, BB * HH), 128, 0, stream>>>(Qb, Kb, Vtb, Yb);

    gemm_bf16_k<3><<<ggrid, 256, 0, stream>>>(Yb, wpT, bp, d_out);
}
